// SiglipEncoderLayer_996432413229
// MI455X (gfx1250) — matrix operand order repaired and checked by reading, not compiled, not run on hardware
//
#include <hip/hip_runtime.h>

// ---------------------------------------------------------------------------
// SigLIP encoder layer for MI455X (gfx1250), wave32 + WMMA f16->f32.
// ~290 GFLOP vs ~150MB traffic => compute-bound: every GEMM uses
// v_wmma_f32_16x16x32_f16. Attention is flash-style (no [B,H,S,S] tensor).
// Tile staging uses the Tensor Data Mover (tensor_load_to_lds + TENSORcnt)
// double-buffered so DMA overlaps WMMA; falls back to per-lane copies if the
// TDM builtin is unavailable.
// ---------------------------------------------------------------------------

typedef __attribute__((ext_vector_type(16))) _Float16 v16h;
typedef __attribute__((ext_vector_type(8)))  float    v8f;

#define E_DIM  1152
#define S_DIM  1024
#define B_DIM  8
#define H_DIM  16
#define HD     72      // head dim
#define HDP    96      // head dim padded to 3x32 for WMMA K-chunks
#define I_DIM  4304
#define I_PAD  4352    // padded to multiple of 128 (zero-filled pad cols)
#define TOKENS (B_DIM * S_DIM)

#if defined(__AMDGCN__) && __has_builtin(__builtin_amdgcn_tensor_load_to_lds) && \
    __has_builtin(__builtin_amdgcn_s_wait_tensorcnt)
#define HAVE_TDM 1
#else
#define HAVE_TDM 0
#endif

#if HAVE_TDM
typedef __attribute__((ext_vector_type(4))) unsigned int u32x4;
typedef __attribute__((ext_vector_type(8))) int          i32x8;
typedef __attribute__((ext_vector_type(4))) int          i32x4;
typedef void __attribute__((address_space(3))) lds_void;

// TDM 2D tile load, data_size = 2B. LDS destination layout: stream of 32-half
// chunks, each padded to 40 halves (pad_interval = 16 DWORDs, pad_amount = 4
// DWORDs). row_halves must be a multiple of 32. Rows beyond valid_rows are
// zero-filled by the TDM OOB logic (tensor_dim1 clamp).
static __device__ __forceinline__ void tdm_load_2d(const _Float16* gsrc,
                                                   _Float16* lds_dst,
                                                   int row_halves, int rows,
                                                   int valid_rows,
                                                   long stride_halves) {
  unsigned long long ga = (unsigned long long)(uintptr_t)gsrc;
  unsigned int ldsa = (unsigned int)(uintptr_t)(lds_void*)(void*)lds_dst;
  u32x4 g0 = {1u,                                  // count=1, is_restore=0
              ldsa,                                // lds_addr
              (unsigned int)ga,                    // global_addr[31:0]
              (unsigned int)((ga >> 32) & 0x1FFFFFFu) | 0x80000000u};  // +type=2
  i32x8 g1 = {
      (int)((1u << 16) |                 // data_size = 2 bytes
            (1u << 20) |                 // pad_enable
            (3u << 22) |                 // pad_interval = 16 DWORDs
            (3u << 25)),                 // pad_amount  = 4 DWORDs
      (int)((row_halves & 0xFFFF) << 16),                    // tensor_dim0 lo16
      (int)(((row_halves >> 16) & 0xFFFF) |
            ((valid_rows & 0xFFFF) << 16)),                  // td0 hi / td1 lo
      (int)(((valid_rows >> 16) & 0xFFFF) |
            ((row_halves & 0xFFFF) << 16)),                  // td1 hi / tile_dim0
      (int)(rows & 0xFFFF),                                  // tile_dim1, dim2=0
      (int)(unsigned int)(stride_halves & 0xFFFFFFFFll),     // dim0_stride lo32
      (int)(unsigned int)((stride_halves >> 32) & 0xFFFFll), // dim0_stride hi16
      0};
  i32x4 z4 = {0, 0, 0, 0};
#if defined(__clang_major__) && __clang_major__ >= 23
  i32x8 z8 = {0, 0, 0, 0, 0, 0, 0, 0};
  __builtin_amdgcn_tensor_load_to_lds(g0, g1, z4, z4, z8, 0);
#else
  __builtin_amdgcn_tensor_load_to_lds(g0, g1, z4, z4, 0);
#endif
}
#endif  // HAVE_TDM

static __device__ __forceinline__ float gelu_tanh(float x) {
  float x3 = x * x * x;
  return 0.5f * x * (1.0f + tanhf(0.79788456080286535588f * (x + 0.044715f * x3)));
}

// xor-butterfly reductions across the 16-lane column groups (wave32) using
// ds_swizzle with immediate xor masks (and_mask=0x1F, or_mask=0, xor=m).
static __device__ __forceinline__ float rmax16(float v) {
  v = fmaxf(v, __int_as_float(__builtin_amdgcn_ds_swizzle(__float_as_int(v), (1 << 10) | 0x1F)));
  v = fmaxf(v, __int_as_float(__builtin_amdgcn_ds_swizzle(__float_as_int(v), (2 << 10) | 0x1F)));
  v = fmaxf(v, __int_as_float(__builtin_amdgcn_ds_swizzle(__float_as_int(v), (4 << 10) | 0x1F)));
  v = fmaxf(v, __int_as_float(__builtin_amdgcn_ds_swizzle(__float_as_int(v), (8 << 10) | 0x1F)));
  return v;
}
static __device__ __forceinline__ float rsum16(float v) {
  v += __int_as_float(__builtin_amdgcn_ds_swizzle(__float_as_int(v), (1 << 10) | 0x1F));
  v += __int_as_float(__builtin_amdgcn_ds_swizzle(__float_as_int(v), (2 << 10) | 0x1F));
  v += __int_as_float(__builtin_amdgcn_ds_swizzle(__float_as_int(v), (4 << 10) | 0x1F));
  v += __int_as_float(__builtin_amdgcn_ds_swizzle(__float_as_int(v), (8 << 10) | 0x1F));
  return v;
}

// A-fragment (16x32 f16) per ISA layout: lane r=lane&15 -> row, lane-half g
// selects K sub-ranges; element j<8 -> k=g*8+j, j>=8 -> k=16+g*8+(j-8).
static __device__ __forceinline__ v16h load_a_frag(const _Float16* base, int ld,
                                                   int row0, int nl, int g) {
  const _Float16* p = base + (size_t)(row0 + nl) * ld;
  v16h a;
#pragma unroll
  for (int j = 0; j < 8; ++j) a[j] = p[g * 8 + j];
#pragma unroll
  for (int j = 0; j < 8; ++j) a[8 + j] = p[16 + g * 8 + j];
  return a;
}

// B-fragment (32x16 f16): lane nl = N column, element j<8 -> k = g*8 + j,
// j>=8 -> k = 16 + g*8 + (j-8). Tile stored transposed in LDS as Bs[n][k].
static __device__ __forceinline__ v16h load_b_frag(const _Float16* base, int ld,
                                                   int n0, int nl, int g) {
  const _Float16* p = base + (size_t)(n0 + nl) * ld;
  v16h b;
#pragma unroll
  for (int j = 0; j < 8; ++j) { b[j] = p[g * 8 + j]; b[8 + j] = p[16 + g * 8 + j]; }
  return b;
}

// ---------------------------------------------------------------------------
// Weight convert + transpose: out[n*Kpad + k] = (f16) w[k*N + n], zero pad k>=K
// ---------------------------------------------------------------------------
__global__ void cvt_transpose_kernel(const float* __restrict__ w,
                                     _Float16* __restrict__ out,
                                     int K, int N, int Kpad) {
  size_t idx = (size_t)blockIdx.x * 256 + threadIdx.x;
  size_t total = (size_t)N * Kpad;
  if (idx >= total) return;
  int n = (int)(idx / Kpad);
  int k = (int)(idx % Kpad);
  out[idx] = (k < K) ? (_Float16)w[(size_t)k * N + n] : (_Float16)0.0f;
}

__global__ void fill_zero_kernel(uint4* __restrict__ p, int n) {
  int i = blockIdx.x * 256 + threadIdx.x;
  if (i < n) p[i] = make_uint4(0u, 0u, 0u, 0u);
}

// ---------------------------------------------------------------------------
// LayerNorm over E=1152, f32 in -> f16 out. One block per token row.
// ---------------------------------------------------------------------------
__global__ __launch_bounds__(256) void ln_kernel(const float* __restrict__ x,
                                                 const float* __restrict__ gg,
                                                 const float* __restrict__ bb,
                                                 _Float16* __restrict__ out) {
  __shared__ float red[256];
  const int row = blockIdx.x, tid = threadIdx.x;
  const float* xr = x + (size_t)row * E_DIM;
  float s = 0.0f, s2 = 0.0f;
  for (int i = tid; i < E_DIM; i += 256) {
    float v = xr[i];
    s += v; s2 += v * v;
  }
  red[tid] = s; __syncthreads();
  for (int st = 128; st > 0; st >>= 1) { if (tid < st) red[tid] += red[tid + st]; __syncthreads(); }
  float mu = red[0] * (1.0f / E_DIM);
  __syncthreads();
  red[tid] = s2; __syncthreads();
  for (int st = 128; st > 0; st >>= 1) { if (tid < st) red[tid] += red[tid + st]; __syncthreads(); }
  float var = red[0] * (1.0f / E_DIM) - mu * mu;
  float rs = rsqrtf(var + 1e-6f);
  for (int i = tid; i < E_DIM; i += 256)
    out[(size_t)row * E_DIM + i] = (_Float16)((xr[i] - mu) * rs * gg[i] + bb[i]);
}

// ---------------------------------------------------------------------------
// Tiled WMMA GEMM: C[M,N] = A[M,K](f16,row,ld=K) @ B[K,N] (stored transposed
// BT[N][ldbt] f16) + bias. 128x128 block, BK=32, 8 waves, each 32x64 (2x4
// WMMA tiles). Double-buffered LDS; TDM DMA staging when available.
// OUTMODE 0: f16 row-major out (ldo), zero-fill cols [N,Npad)
// OUTMODE 1: f16 QKV pack to [B,H,S,HDP]
// OUTMODE 2: f32 out = acc + bias + residual[row*N+col]
// ---------------------------------------------------------------------------
template <int OUTMODE, bool GELU>
__global__ __launch_bounds__(256) void gemm_wmma_kernel(
    const _Float16* __restrict__ A, const _Float16* __restrict__ BT,
    const float* __restrict__ bias, const float* __restrict__ residual,
    void* __restrict__ out, int N, int Npad, int K, int ldbt, int ldo) {
  __shared__ __align__(16) _Float16 As[2][128 * 40];
  __shared__ __align__(16) _Float16 Bs[2][128 * 40];
  const int tid = threadIdx.x;
  const int w = tid >> 5, lane = tid & 31;
  const int nl = lane & 15, g = lane >> 4;
  const int wm = (w >> 1) * 32, wn = (w & 1) * 64;
  const int tn0 = blockIdx.x * 128;
  const int tm0 = blockIdx.y * 128;
  const int bvalid = (N - tn0 < 128) ? (N - tn0) : 128;
  (void)bvalid;

  v8f acc[2][4] = {};

#if HAVE_TDM
#define GEMM_STAGE(buf, kk)                                                   \
  do {                                                                        \
    if (w == 0) {                                                             \
      tdm_load_2d(A + (size_t)tm0 * K + (kk), As[buf], 32, 128, 128, K);      \
      tdm_load_2d(BT + (size_t)tn0 * ldbt + (kk), Bs[buf], 32, 128, bvalid,   \
                  ldbt);                                                      \
    }                                                                         \
  } while (0)
#define GEMM_STAGE_WAIT()                                                     \
  do {                                                                        \
    if (w == 0) __builtin_amdgcn_s_wait_tensorcnt(0);                         \
  } while (0)
#else
#define GEMM_STAGE(buf, kk)                                                   \
  do {                                                                        \
    for (int rep = 0; rep < 2; ++rep) {                                       \
      int u = tid + rep * 256;                                                \
      int r = u >> 2, seg = u & 3;                                            \
      *(uint4*)(As[buf] + r * 40 + seg * 8) =                                 \
          *(const uint4*)(A + (size_t)(tm0 + r) * K + (kk) + seg * 8);        \
      uint4 bvv;                                                              \
      if (tn0 + r < N)                                                        \
        bvv = *(const uint4*)(BT + (size_t)(tn0 + r) * ldbt + (kk) + seg * 8);\
      else                                                                    \
        bvv = make_uint4(0u, 0u, 0u, 0u);                                     \
      *(uint4*)(Bs[buf] + r * 40 + seg * 8) = bvv;                            \
    }                                                                         \
  } while (0)
#define GEMM_STAGE_WAIT() do { } while (0)
#endif

  const int nsteps = K >> 5;
  GEMM_STAGE(0, 0);
  int cur = 0;
  for (int s = 0; s < nsteps; ++s) {
    GEMM_STAGE_WAIT();
    __syncthreads();
    if (s + 1 < nsteps) GEMM_STAGE(cur ^ 1, (s + 1) * 32);  // overlap DMA
    const _Float16* as = As[cur];
    const _Float16* bs = Bs[cur];
    v16h af[2], bf[4];
#pragma unroll
    for (int mt = 0; mt < 2; ++mt) af[mt] = load_a_frag(as, 40, wm + mt * 16, nl, g);
#pragma unroll
    for (int nt = 0; nt < 4; ++nt) bf[nt] = load_b_frag(bs, 40, wn + nt * 16, nl, g);
#pragma unroll
    for (int mt = 0; mt < 2; ++mt)
#pragma unroll
      for (int nt = 0; nt < 4; ++nt)
        acc[mt][nt] = __builtin_amdgcn_wmma_f32_16x16x32_f16(
            false, af[mt], false, bf[nt], (short)0, acc[mt][nt], false, false);
    cur ^= 1;
  }
#undef GEMM_STAGE
#undef GEMM_STAGE_WAIT

  // epilogue (C layout: VGPR i -> row g*8+i, lane nl -> col)
#pragma unroll
  for (int mt = 0; mt < 2; ++mt) {
#pragma unroll
    for (int nt = 0; nt < 4; ++nt) {
      int col = tn0 + wn + nt * 16 + nl;
      float bb = (col < N) ? bias[col] : 0.0f;
#pragma unroll
      for (int i = 0; i < 8; ++i) {
        int row = tm0 + wm + mt * 16 + g * 8 + i;
        float v = acc[mt][nt][i] + bb;
        if (GELU) v = gelu_tanh(v);
        if (OUTMODE == 0) {
          _Float16* o = (_Float16*)out;
          if (col < N)          o[(size_t)row * ldo + col] = (_Float16)v;
          else if (col < Npad)  o[(size_t)row * ldo + col] = (_Float16)0.0f;
        } else if (OUTMODE == 1) {
          _Float16* o = (_Float16*)out;
          int h = col / HD, d = col - h * HD;
          int b = row >> 10, s = row & (S_DIM - 1);
          o[(((size_t)(b * H_DIM + h)) * S_DIM + s) * HDP + d] = (_Float16)v;
        } else {
          if (col < N) {
            float r = residual[(size_t)row * N + col];
            ((float*)out)[(size_t)row * ldo + col] = v + r;
          }
        }
      }
    }
  }
}

// ---------------------------------------------------------------------------
// Flash attention: grid = B*H * (S/128); 8 waves, each wave owns 16 q rows.
// Q/K/V are f16 [B,H,S,HDP] (d>=72 zero-padded on the Q side so dot products
// over the pad are exact zeros). QK^T and P*V both use WMMA; K tile staged by
// TDM (chunk-padded layout), V staged transposed, online softmax row stats
// reduced with ds_swizzle xor butterflies.
// ---------------------------------------------------------------------------
__global__ __launch_bounds__(256) void attn_kernel(const _Float16* __restrict__ Q,
                                                   const _Float16* __restrict__ Km,
                                                   const _Float16* __restrict__ V,
                                                   _Float16* __restrict__ O) {
  // ktile: 32 keys x 96 d, stored as 32-half chunks padded to 40 halves:
  // element (key, c*32+t) at ktile[(key*3 + c)*40 + t]
  __shared__ __align__(16) _Float16 ktile[32 * 3 * 40];
  __shared__ __align__(16) _Float16 vt[96 * 40];       // [d][key] transposed
  __shared__ __align__(16) _Float16 pl[8 * 16 * 40];   // per-wave P staging
  const int tid = threadIdx.x;
  const int w = tid >> 5, lane = tid & 31;
  const int nl = lane & 15, g = lane >> 4;
  const int qblk = blockIdx.x & 7;
  const int bh = blockIdx.x >> 3;
  const _Float16* Qb = Q + (size_t)bh * S_DIM * HDP;
  const _Float16* Kb = Km + (size_t)bh * S_DIM * HDP;
  const _Float16* Vb = V + (size_t)bh * S_DIM * HDP;
  const int qbase = qblk * 128 + w * 16;
  _Float16* plw = pl + w * (16 * 40);

  // Q fragments, 3 K-chunks of 32 over padded head dim (held in registers)
  v16h qf[3];
#pragma unroll
  for (int c = 0; c < 3; ++c) {
    const _Float16* p = Qb + (size_t)(qbase + nl) * HDP + c * 32;
#pragma unroll
    for (int j = 0; j < 8; ++j) { qf[c][j] = p[g * 8 + j]; qf[c][8 + j] = p[16 + g * 8 + j]; }
  }

  float mrow[8], lrow[8];
#pragma unroll
  for (int i = 0; i < 8; ++i) { mrow[i] = -1e30f; lrow[i] = 0.0f; }
  v8f oacc[6] = {};
  const float scale = 0.11785113019775793f;  // 72^-0.5

  for (int kb = 0; kb < S_DIM; kb += 32) {
    __syncthreads();  // protect ktile/vt/pl overwrite vs previous iter's reads
#if HAVE_TDM
    if (w == 0) tdm_load_2d(Kb + (size_t)kb * HDP, ktile, 96, 32, 32, HDP);
#else
    for (int u = tid; u < 384; u += 256) {
      int key = u / 12, seg = u % 12;
      *(uint4*)(ktile + (key * 3 + (seg >> 2)) * 40 + (seg & 3) * 8) =
          *(const uint4*)(Kb + (size_t)(kb + key) * HDP + seg * 8);
    }
#endif
    // V tile transposed (scalar scatter; TDM cannot transpose)
    for (int u = tid; u < 384; u += 256) {
      int key = u / 12, seg = u % 12;
      const _Float16* src = Vb + (size_t)(kb + key) * HDP + seg * 8;
#pragma unroll
      for (int e = 0; e < 8; ++e) vt[(seg * 8 + e) * 40 + key] = src[e];
    }
#if HAVE_TDM
    if (w == 0) __builtin_amdgcn_s_wait_tensorcnt(0);
#endif
    __syncthreads();

    // scores: 2 key-subtiles x 3 d-chunks of WMMA
    v8f sc[2] = {};
#pragma unroll
    for (int nt = 0; nt < 2; ++nt) {
#pragma unroll
      for (int c = 0; c < 3; ++c) {
        v16h kf;
        const _Float16* p = ktile + ((nt * 16 + nl) * 3 + c) * 40;
#pragma unroll
        for (int j = 0; j < 8; ++j) { kf[j] = p[g * 8 + j]; kf[8 + j] = p[16 + g * 8 + j]; }
        sc[nt] = __builtin_amdgcn_wmma_f32_16x16x32_f16(
            false, qf[c], false, kf, (short)0, sc[nt], false, false);
      }
    }

    // online softmax (rows = VGPR index within 16-lane group)
#pragma unroll
    for (int i = 0; i < 8; ++i) {
      float s0 = sc[0][i] * scale, s1 = sc[1][i] * scale;
      float mx = rmax16(fmaxf(s0, s1));
      float mnew = fmaxf(mrow[i], mx);
      float alpha = __expf(mrow[i] - mnew);
      float p0 = __expf(s0 - mnew), p1 = __expf(s1 - mnew);
      float rs = rsum16(p0 + p1);
      lrow[i] = lrow[i] * alpha + rs;
      mrow[i] = mnew;
#pragma unroll
      for (int dt = 0; dt < 6; ++dt) oacc[dt][i] *= alpha;
      plw[(g * 8 + i) * 40 + nl] = (_Float16)p0;         // C-layout -> LDS
      plw[(g * 8 + i) * 40 + 16 + nl] = (_Float16)p1;
    }
    __syncthreads();

    // reload P in A-layout, accumulate P @ V over 6 d-tiles
    v16h pf;
    {
      const _Float16* p = plw + nl * 40;
#pragma unroll
      for (int j = 0; j < 8; ++j) { pf[j] = p[g * 8 + j]; pf[8 + j] = p[16 + g * 8 + j]; }
    }
#pragma unroll
    for (int dt = 0; dt < 6; ++dt) {
      v16h vf;
      const _Float16* p = vt + (dt * 16 + nl) * 40;
#pragma unroll
      for (int j = 0; j < 8; ++j) { vf[j] = p[g * 8 + j]; vf[8 + j] = p[16 + g * 8 + j]; }
      oacc[dt] = __builtin_amdgcn_wmma_f32_16x16x32_f16(
          false, pf, false, vf, (short)0, oacc[dt], false, false);
    }
  }

  // normalize and store head-concat f16 [T, E]
  const int b = bh >> 4, h = bh & 15;
#pragma unroll
  for (int dt = 0; dt < 6; ++dt) {
    int d = dt * 16 + nl;
    if (d < HD) {
#pragma unroll
      for (int i = 0; i < 8; ++i) {
        int srow = qbase + g * 8 + i;
        float val = oacc[dt][i] / lrow[i];
        O[((size_t)(b * S_DIM + srow)) * E_DIM + h * HD + d] = (_Float16)val;
      }
    }
  }
}

// ---------------------------------------------------------------------------
// Host-side launch sequence
// ---------------------------------------------------------------------------
extern "C" void kernel_launch(void* const* d_in, const int* in_sizes, int n_in,
                              void* d_out, int out_size, void* d_ws, size_t ws_size,
                              hipStream_t stream) {
  const float* hidden = (const float*)d_in[0];
  const float* wq = (const float*)d_in[1];  const float* bq = (const float*)d_in[2];
  const float* wk = (const float*)d_in[3];  const float* bk = (const float*)d_in[4];
  const float* wv = (const float*)d_in[5];  const float* bv = (const float*)d_in[6];
  const float* wo = (const float*)d_in[7];  const float* bo = (const float*)d_in[8];
  const float* ln1g = (const float*)d_in[9];  const float* ln1b = (const float*)d_in[10];
  const float* ln2g = (const float*)d_in[11]; const float* ln2b = (const float*)d_in[12];
  const float* w1 = (const float*)d_in[13]; const float* b1 = (const float*)d_in[14];
  const float* w2 = (const float*)d_in[15]; const float* b2 = (const float*)d_in[16];
  float* out = (float*)d_out;

  // workspace carve (~272 MB)
  char* ws = (char*)d_ws;
  size_t off = 0;
  auto carve = [&](size_t bytes) -> char* {
    char* p = ws + off;
    off += (bytes + 255) & ~(size_t)255;
    return p;
  };
  _Float16* wqT = (_Float16*)carve((size_t)E_DIM * E_DIM * 2);
  _Float16* wkT = (_Float16*)carve((size_t)E_DIM * E_DIM * 2);
  _Float16* wvT = (_Float16*)carve((size_t)E_DIM * E_DIM * 2);
  _Float16* woT = (_Float16*)carve((size_t)E_DIM * E_DIM * 2);
  _Float16* w1T = (_Float16*)carve((size_t)I_DIM * E_DIM * 2);   // [I][E]
  _Float16* w2T = (_Float16*)carve((size_t)E_DIM * I_PAD * 2);   // [E][I_PAD]
  _Float16* xln1 = (_Float16*)carve((size_t)TOKENS * E_DIM * 2);
  _Float16* qp = (_Float16*)carve((size_t)B_DIM * H_DIM * S_DIM * HDP * 2);
  _Float16* kp = (_Float16*)carve((size_t)B_DIM * H_DIM * S_DIM * HDP * 2);
  _Float16* vp = (_Float16*)carve((size_t)B_DIM * H_DIM * S_DIM * HDP * 2);
  _Float16* attn16 = (_Float16*)carve((size_t)TOKENS * E_DIM * 2);
  float*    x1 = (float*)carve((size_t)TOKENS * E_DIM * 4);
  _Float16* xln2 = (_Float16*)carve((size_t)TOKENS * E_DIM * 2);
  _Float16* h1 = (_Float16*)carve((size_t)TOKENS * I_PAD * 2);

  // 1) weight convert + transpose (f32 -> f16, [N][Kpad])
  {
    int tot = E_DIM * E_DIM;
    int gb = (tot + 255) / 256;
    cvt_transpose_kernel<<<gb, 256, 0, stream>>>(wq, wqT, E_DIM, E_DIM, E_DIM);
    cvt_transpose_kernel<<<gb, 256, 0, stream>>>(wk, wkT, E_DIM, E_DIM, E_DIM);
    cvt_transpose_kernel<<<gb, 256, 0, stream>>>(wv, wvT, E_DIM, E_DIM, E_DIM);
    cvt_transpose_kernel<<<gb, 256, 0, stream>>>(wo, woT, E_DIM, E_DIM, E_DIM);
    int tot1 = I_DIM * E_DIM;
    cvt_transpose_kernel<<<(tot1 + 255) / 256, 256, 0, stream>>>(w1, w1T, E_DIM, I_DIM, E_DIM);
    int tot2 = E_DIM * I_PAD;
    cvt_transpose_kernel<<<(tot2 + 255) / 256, 256, 0, stream>>>(w2, w2T, I_DIM, E_DIM, I_PAD);
  }

  // 2) zero QKV padded buffers (so d>=72 pad lanes of Q are exact zeros)
  {
    int nU4 = (B_DIM * H_DIM * S_DIM * HDP * 2) / 16;
    fill_zero_kernel<<<(nU4 + 255) / 256, 256, 0, stream>>>((uint4*)qp, nU4);
    fill_zero_kernel<<<(nU4 + 255) / 256, 256, 0, stream>>>((uint4*)kp, nU4);
    fill_zero_kernel<<<(nU4 + 255) / 256, 256, 0, stream>>>((uint4*)vp, nU4);
  }

  // 3) LN1
  ln_kernel<<<TOKENS, 256, 0, stream>>>(hidden, ln1g, ln1b, xln1);

  // 4) QKV projections -> packed [B,H,S,HDP] f16
  dim3 gE((E_DIM + 127) / 128, TOKENS / 128);
  gemm_wmma_kernel<1, false><<<gE, 256, 0, stream>>>(xln1, wqT, bq, nullptr, qp,
                                                     E_DIM, E_DIM, E_DIM, E_DIM, 0);
  gemm_wmma_kernel<1, false><<<gE, 256, 0, stream>>>(xln1, wkT, bk, nullptr, kp,
                                                     E_DIM, E_DIM, E_DIM, E_DIM, 0);
  gemm_wmma_kernel<1, false><<<gE, 256, 0, stream>>>(xln1, wvT, bv, nullptr, vp,
                                                     E_DIM, E_DIM, E_DIM, E_DIM, 0);

  // 5) flash attention
  attn_kernel<<<B_DIM * H_DIM * (S_DIM / 128), 256, 0, stream>>>(qp, kp, vp, attn16);

  // 6) output projection + residual -> x1 (f32)
  gemm_wmma_kernel<2, false><<<gE, 256, 0, stream>>>(attn16, woT, bo, hidden, x1,
                                                     E_DIM, E_DIM, E_DIM, E_DIM, E_DIM);

  // 7) LN2
  ln_kernel<<<TOKENS, 256, 0, stream>>>(x1, ln2g, ln2b, xln2);

  // 8) MLP up + GELU -> h1 f16 [T, I_PAD] (pad cols zero-stored in epilogue)
  dim3 gI((I_PAD + 127) / 128, TOKENS / 128);
  gemm_wmma_kernel<0, true><<<gI, 256, 0, stream>>>(xln2, w1T, b1, nullptr, h1,
                                                    I_DIM, I_PAD, E_DIM, E_DIM, I_PAD);

  // 9) MLP down + residual -> d_out f32 (K = I_PAD, pad K rows of w2T are zero)
  gemm_wmma_kernel<2, false><<<gE, 256, 0, stream>>>(h1, w2T, b2, x1, out,
                                                     E_DIM, E_DIM, I_PAD, I_PAD, E_DIM);
}
